// StructureAE_11828339933654
// MI455X (gfx1250) — compile-verified, matrix operand order checked by reading, not run on hardware
//
#include <hip/hip_runtime.h>
#include <hip/hip_bf16.h>

typedef __attribute__((ext_vector_type(16))) _Float16     v16h;
typedef __attribute__((ext_vector_type(8)))  float        v8f;
typedef __attribute__((ext_vector_type(4)))  unsigned int v4u;
typedef __attribute__((ext_vector_type(8)))  unsigned int v8u;

#define NN   10000
#define IND  128
#define EMB  128
#define HID  64
#define NEG_SLOPE 0.2f

// ---------- helpers ----------
__device__ __forceinline__ unsigned f2ord(float f) {
    unsigned u = __float_as_uint(f);
    return (u & 0x80000000u) ? ~u : (u | 0x80000000u);
}
__device__ __forceinline__ float ord2f(unsigned u) {
    return (u & 0x80000000u) ? __uint_as_float(u & 0x7fffffffu)
                             : __uint_as_float(~u);
}
__device__ __forceinline__ float fast_sigmoid(float v) {
    return 1.0f / (1.0f + __expf(-v));
}

// =====================================================================
// Kernel 1: fused  h = relu(x@W1+b1);  z = h@W2;  a_src=z.att_s; a_dst=z.att_d
// One wave (32 threads) per 16 rows.  GEMMs via v_wmma_f32_16x16x32_f16.
// A-frag: lane(0..15)=row M, VGPR v holds K = (v&3)*2 + (lane>>4)*8 + (v>>2)*16 (+1)
// B-frag: lane(0..15)=col N, same K pattern.
// C/D   : VGPR v -> M = v + (lane>>4)*8, N = lane&15.
// =====================================================================
__global__ __launch_bounds__(32) void k_dense_gat(
    const float* __restrict__ x, const float* __restrict__ dense_w,
    const float* __restrict__ dense_b, const float* __restrict__ gat_w,
    const float* __restrict__ att_s, const float* __restrict__ att_d,
    float* __restrict__ z, float* __restrict__ a_src, float* __restrict__ a_dst)
{
    __shared__ _Float16 h_lds[16 * EMB];   // relu(x@W1+b1) tile, f16
    __shared__ float    z_lds[16 * HID];

    const int lane = threadIdx.x;
    const int r    = lane & 15;
    const int kh   = lane >> 4;
    const int row0 = blockIdx.x * 16;

    // ---- stage 1: h tile (16 x 128), accumulate over K=128 ----
    for (int nt = 0; nt < EMB / 16; ++nt) {
        v8f c = {};
        for (int k0 = 0; k0 < IND; k0 += 32) {
            v16h a, b;
#pragma unroll
            for (int v = 0; v < 8; ++v) {
                const int ka = k0 + (v & 3) * 2 + kh * 8 + (v >> 2) * 16;
                a[2 * v]     = (_Float16)x[(row0 + r) * IND + ka];
                a[2 * v + 1] = (_Float16)x[(row0 + r) * IND + ka + 1];
                b[2 * v]     = (_Float16)dense_w[ka * EMB + nt * 16 + r];
                b[2 * v + 1] = (_Float16)dense_w[(ka + 1) * EMB + nt * 16 + r];
            }
            c = __builtin_amdgcn_wmma_f32_16x16x32_f16(false, a, false, b,
                                                       (short)0, c, false, false);
        }
#pragma unroll
        for (int v = 0; v < 8; ++v) {
            const int m = v + kh * 8;
            float val = c[v] + dense_b[nt * 16 + r];
            h_lds[m * EMB + nt * 16 + r] = (_Float16)fmaxf(val, 0.0f);
        }
    }
    __syncthreads();

    // ---- stage 2: z tile (16 x 64) = h_tile @ gat_w ----
    for (int nt = 0; nt < HID / 16; ++nt) {
        v8f c = {};
        for (int k0 = 0; k0 < EMB; k0 += 32) {
            v16h a, b;
#pragma unroll
            for (int v = 0; v < 8; ++v) {
                const int ka = k0 + (v & 3) * 2 + kh * 8 + (v >> 2) * 16;
                a[2 * v]     = h_lds[r * EMB + ka];
                a[2 * v + 1] = h_lds[r * EMB + ka + 1];
                b[2 * v]     = (_Float16)gat_w[ka * HID + nt * 16 + r];
                b[2 * v + 1] = (_Float16)gat_w[(ka + 1) * HID + nt * 16 + r];
            }
            c = __builtin_amdgcn_wmma_f32_16x16x32_f16(false, a, false, b,
                                                       (short)0, c, false, false);
        }
#pragma unroll
        for (int v = 0; v < 8; ++v)
            z_lds[(v + kh * 8) * HID + nt * 16 + r] = c[v];
    }
    __syncthreads();

    // write z, and per-row attention dots (lanes 0-15: src, 16-31: dst)
    for (int i = lane; i < 16 * HID; i += 32)
        z[(size_t)row0 * HID + i] = z_lds[i];

    {
        const float* att = (lane < 16) ? att_s : att_d;
        float s = 0.0f;
#pragma unroll 8
        for (int k = 0; k < HID; ++k) s += z_lds[r * HID + k] * att[k];
        if (lane < 16) a_src[row0 + r] = s;
        else           a_dst[row0 + r] = s;
    }
}

// =====================================================================
// Segment softmax + scatter pipeline
// =====================================================================
__global__ void k_init(float* __restrict__ segsum, unsigned* __restrict__ segmax,
                       float* __restrict__ emb)
{
    int i = blockIdx.x * blockDim.x + threadIdx.x;
    if (i < NN) { segsum[i] = 0.0f; segmax[i] = 0u; }   // 0u == ordered(-inf-ish)
    if (i < NN * HID) emb[i] = 0.0f;
}

__global__ void k_edge_max(const int* __restrict__ ei,
                           const float* __restrict__ asrc,
                           const float* __restrict__ adst,
                           unsigned* __restrict__ segmax, int E_)
{
    int i = blockIdx.x * blockDim.x + threadIdx.x;
    if (i >= E_ + NN) return;
    int s, d;
    if (i < E_) { s = ei[i]; d = ei[E_ + i]; } else { s = d = i - E_; }
    float e = asrc[s] + adst[d];
    e = (e > 0.0f) ? e : NEG_SLOPE * e;
    atomicMax(&segmax[d], f2ord(e));
}

__global__ void k_edge_sum(const int* __restrict__ ei,
                           const float* __restrict__ asrc,
                           const float* __restrict__ adst,
                           const unsigned* __restrict__ segmax,
                           float* __restrict__ segsum,
                           float* __restrict__ exbuf, int E_)
{
    int i = blockIdx.x * blockDim.x + threadIdx.x;
    if (i >= E_ + NN) return;
    int s, d;
    if (i < E_) { s = ei[i]; d = ei[E_ + i]; } else { s = d = i - E_; }
    float e = asrc[s] + adst[d];
    e = (e > 0.0f) ? e : NEG_SLOPE * e;
    float ex = __expf(e - ord2f(segmax[d]));
    exbuf[i] = ex;
    atomicAdd(&segsum[d], ex);
}

// one wave per edge: lanes cover the 64 hidden channels (2 per lane)
__global__ __launch_bounds__(256) void k_edge_accum(
    const int* __restrict__ ei, const float* __restrict__ exbuf,
    const float* __restrict__ segsum, const float* __restrict__ z,
    float* __restrict__ emb, int E_)
{
    int widx = blockIdx.x * (blockDim.x >> 5) + (threadIdx.x >> 5);
    int lane = threadIdx.x & 31;
    if (widx >= E_ + NN) return;
    int s, d;
    if (widx < E_) { s = ei[widx]; d = ei[E_ + widx]; } else { s = d = widx - E_; }
    float alpha = exbuf[widx] / segsum[d];
    float v0 = alpha * z[(size_t)s * HID + lane];
    float v1 = alpha * z[(size_t)s * HID + lane + 32];
    atomicAdd(&emb[(size_t)d * HID + lane], v0);
    atomicAdd(&emb[(size_t)d * HID + lane + 32], v1);
}

// add bias, emit f32 embed output + f16 copy for the sim GEMM
__global__ void k_fin(const float* __restrict__ emb, const float* __restrict__ gb,
                      float* __restrict__ out_emb, _Float16* __restrict__ zh)
{
    int i = blockIdx.x * blockDim.x + threadIdx.x;
    if (i >= NN * HID) return;
    float v = emb[i] + gb[i & (HID - 1)];
    out_emb[i] = v;
    zh[i] = (_Float16)v;
}

// =====================================================================
// Kernel: sim = sigmoid(embed @ embed^T)
// Block = 8 waves; block covers row-tile (16 rows) x 8 col-tiles.
// The shared 16x64 f16 A-tile (2KB, contiguous in zh) is staged into LDS
// once per block by the Tensor Data Mover (tensor_load_to_lds, TENSORcnt),
// then each wave runs K=64 in 2 WMMA steps with a fused sigmoid epilogue.
// =====================================================================
__global__ __launch_bounds__(256) void k_sim(const _Float16* __restrict__ zh,
                                             float* __restrict__ sim)
{
    __shared__ _Float16 a_tile[16 * HID];   // 2 KB row tile

    const int lane = threadIdx.x & 31;
    const int wave = threadIdx.x >> 5;
    const int r    = lane & 15;
    const int kh   = lane >> 4;

    const int row0 = blockIdx.x * 16;
    const int col0 = (blockIdx.y * 8 + wave) * 16;
    const bool active = (col0 < NN);        // wave-uniform (EXEC stays all-1s)

    // ---- TDM: stage A-tile (2048B = 1 row x 256 x 8B) into LDS ----
    if (wave == 0) {
        unsigned long long ga =
            (unsigned long long)(const void*)(zh + (size_t)row0 * HID);
        unsigned lds_off = (unsigned)(unsigned long long)(uintptr_t)a_tile;

        v4u g0;
        g0[0] = 1u;                                    // count=1 (valid user D#)
        g0[1] = lds_off;                               // lds_addr (bytes)
        g0[2] = (unsigned)(ga & 0xFFFFFFFFu);          // global_addr[31:0]
        g0[3] = (unsigned)((ga >> 32) & 0x01FFFFFFu)   // global_addr[56:32]
              | (2u << 30);                            // type=2 ("image")

        v8u g1;
        g1[0] = (3u << 16);          // workgroup_mask=0 | data_size=3 (8B)
        g1[1] = (256u << 16);        // tensor_dim0[15:0]=256 (bits 79:48 lo)
        g1[2] = (1u << 16);          // tensor_dim0 hi=0 | tensor_dim1=1 lo
        g1[3] = (256u << 16);        // tensor_dim1 hi=0 | tile_dim0=256
        g1[4] = 1u;                  // tile_dim1=1, tile_dim2=0
        g1[5] = 256u;                // tensor_dim0_stride lo
        g1[6] = 0u;                  // stride hi | tensor_dim1_stride lo
        g1[7] = 0u;

        asm volatile("tensor_load_to_lds %0, %1" :: "s"(g0), "s"(g1) : "memory");
        __builtin_amdgcn_s_wait_tensorcnt(0);
    }
    __syncthreads();

    if (active)
        __builtin_prefetch(&zh[(size_t)(col0 + r) * HID], 0, 3); // global_prefetch_b8

    v8f c = {};
    if (active) {
#pragma unroll
        for (int k0 = 0; k0 < HID; k0 += 32) {
            v16h a, b;
#pragma unroll
            for (int v = 0; v < 8; ++v) {
                const int ka = k0 + (v & 3) * 2 + kh * 8 + (v >> 2) * 16;
                a[2 * v]     = a_tile[r * HID + ka];          // LDS (shared A)
                a[2 * v + 1] = a_tile[r * HID + ka + 1];
                b[2 * v]     = zh[(size_t)(col0 + r) * HID + ka];   // L2-resident
                b[2 * v + 1] = zh[(size_t)(col0 + r) * HID + ka + 1];
            }
            c = __builtin_amdgcn_wmma_f32_16x16x32_f16(false, a, false, b,
                                                       (short)0, c, false, false);
        }
#pragma unroll
        for (int v = 0; v < 8; ++v) {
            const int m = row0 + v + kh * 8;
            sim[(size_t)m * NN + col0 + r] = fast_sigmoid(c[v]);
        }
    }
}

// =====================================================================
extern "C" void kernel_launch(void* const* d_in, const int* in_sizes, int n_in,
                              void* d_out, int out_size, void* d_ws, size_t ws_size,
                              hipStream_t stream)
{
    const float* x       = (const float*)d_in[0];
    const int*   ei      = (const int*)  d_in[1];
    const float* dense_w = (const float*)d_in[2];
    const float* dense_b = (const float*)d_in[3];
    const float* gat_w   = (const float*)d_in[4];
    const float* att_s   = (const float*)d_in[5];
    const float* att_d   = (const float*)d_in[6];
    const float* gat_b   = (const float*)d_in[7];

    const int E_ = in_sizes[1] / 2;
    const int NE = E_ + NN;

    // workspace layout (float units)
    float* ws = (float*)d_ws;
    float*    z      = ws;                         // NN*HID
    float*    a_src  = z + (size_t)NN * HID;       // NN
    float*    a_dst  = a_src + NN;                 // NN
    unsigned* segmax = (unsigned*)(a_dst + NN);    // NN
    float*    segsum = (float*)(segmax + NN);      // NN
    float*    emb    = segsum + NN;                // NN*HID
    float*    exbuf  = emb + (size_t)NN * HID;     // E_+NN
    _Float16* zh     = (_Float16*)(exbuf + NE);    // NN*HID halves

    float* sim     = (float*)d_out;
    float* out_emb = sim + (size_t)NN * NN;

    // 1) fused dense+relu -> GAT linear -> attention dots
    k_dense_gat<<<NN / 16, 32, 0, stream>>>(x, dense_w, dense_b, gat_w,
                                            att_s, att_d, z, a_src, a_dst);
    // 2) init segment buffers
    k_init<<<(NN * HID + 255) / 256, 256, 0, stream>>>(segsum, segmax, emb);
    // 3) segment max
    k_edge_max<<<(NE + 255) / 256, 256, 0, stream>>>(ei, a_src, a_dst, segmax, E_);
    // 4) exp + segment sum
    k_edge_sum<<<(NE + 255) / 256, 256, 0, stream>>>(ei, a_src, a_dst, segmax,
                                                     segsum, exbuf, E_);
    // 5) weighted scatter-sum of z into emb
    k_edge_accum<<<(NE + 7) / 8, 256, 0, stream>>>(ei, exbuf, segsum, z, emb, E_);
    // 6) bias + emit embed (f32 tail of d_out) + f16 copy
    k_fin<<<(NN * HID + 255) / 256, 256, 0, stream>>>(emb, gat_b, out_emb, zh);
    // 7) pairwise sigmoid similarity (the big one)
    k_sim<<<dim3(NN / 16, (NN / 16 + 7) / 8), 256, 0, stream>>>(zh, sim);
}
